// TriangleAttention_66589172957486
// MI455X (gfx1250) — compile-verified, matrix operand order checked by reading, not run on hardware
//
#include <hip/hip_runtime.h>

// ---------------------------------------------------------------------------
// Triangle attention for MI455X (gfx1250, wave32, WMMA + async-to-LDS).
// All matmuls use v_wmma_f32_16x16x32_f16 (f16 operands, f32 accumulate).
// Workspace use (~97.2 MB, everything L2-resident on a 192MB L2 part):
//   hn   f16 [65536][128]   16 MB
//   qkvg f16 4x[65536][128] 64 MB   (Q pre-scaled by DH^-0.5)
//   bias f32 [4][65536]      1 MB
//   og   f16 [65536][128]   16 MB   (gated attention output, head-major)
//   wcat f16 [528][128]     132 KB  (q|k|v|g|bias_pad16 rows)
//   wo   f16 [128][128]      32 KB
// ---------------------------------------------------------------------------

typedef __attribute__((ext_vector_type(16))) _Float16 v16h;
typedef __attribute__((ext_vector_type(8)))  float    v8f;

constexpr int NN_ = 65536;   // N*N rows
constexpr int D_  = 128;
constexpr int H_  = 4;

union FragH { v16h h; uint32_t u[8]; };

// Generic->LDS byte address: generic LDS addresses carry the LDS byte offset
// in the low 32 bits.
__device__ inline uint32_t lds_addr_of(const void* p) {
    return (uint32_t)(uintptr_t)p;
}

// Async global->LDS 16-byte copy (GLOBAL_LOAD_ASYNC_TO_LDS_B128, ASYNCcnt).
__device__ inline void async_copy_b128(uint32_t lds_byte_addr, const void* gsrc) {
    asm volatile("global_load_async_to_lds_b128 %0, %1, off"
                 :: "v"(lds_byte_addr), "v"((uint64_t)(uintptr_t)gsrc)
                 : "memory");
}

__device__ inline void wait_async0() {
#if __has_builtin(__builtin_amdgcn_s_wait_asynccnt)
    __builtin_amdgcn_s_wait_asynccnt(0);
#else
    asm volatile("s_wait_asynccnt 0x0" ::: "memory");
#endif
}

// Load a 16x32 f16 A-fragment (or 32x16 B-fragment) per ISA 7.12.2 layout.
// For A: row = lane&15, data at base + row*stride + K.
// For B: col = lane&15, column's K-vector contiguous at base + col*stride + K.
// K interleave: lanes 0-15 -> K in {0..7, 16..23}; lanes 16-31 -> {8..15, 24..31}.
__device__ inline v16h load_frag16(const _Float16* base, int stride, int lane) {
    int rc = lane & 15;
    int hi = (lane >> 4) & 1;
    const _Float16* p = base + rc * stride;
    FragH f;
#pragma unroll
    for (int v = 0; v < 8; ++v) {
        int k = ((v & 4) << 2) + hi * 8 + (v & 3) * 2;   // pairs of consecutive K
        f.u[v] = *(const uint32_t*)(p + k);
    }
    return f.h;
}

// ---------------------------------------------------------------------------
// Kernel 0: weight preconversion to f16 (scale folded into q_w).
// wcat rows: [0..127]=q*scale, [128..255]=k, [256..383]=v, [384..511]=g,
//            [512..515]=bias_w, [516..527]=0
// ---------------------------------------------------------------------------
__global__ void prep_weights(const float* qw, const float* kw, const float* vw,
                             const float* gw, const float* bw, const float* ow,
                             _Float16* wcat, _Float16* wo) {
    int idx = blockIdx.x * 256 + threadIdx.x;
    const float scale = 0.17677669529663687f;   // 1/sqrt(32)
    if (idx < 528 * 128) {
        int r = idx >> 7, c = idx & 127;
        float v;
        if      (r < 128) v = qw[idx] * scale;
        else if (r < 256) v = kw[(r - 128) * 128 + c];
        else if (r < 384) v = vw[(r - 256) * 128 + c];
        else if (r < 512) v = gw[(r - 384) * 128 + c];
        else if (r < 516) v = bw[(r - 512) * 128 + c];
        else              v = 0.0f;
        wcat[idx] = (_Float16)v;
    } else if (idx < 528 * 128 + 128 * 128) {
        int j = idx - 528 * 128;
        wo[j] = (_Float16)ow[j];
    }
}

// ---------------------------------------------------------------------------
// Kernel 1: LayerNorm over D=128. One wave per row: float4 per lane,
// shfl_xor reductions, packed 8-byte f16 store. No LDS, no barriers.
// ---------------------------------------------------------------------------
__global__ __launch_bounds__(256) void ln_kernel(const float* __restrict__ x,
                                                 const float* __restrict__ lnw,
                                                 const float* __restrict__ lnb,
                                                 _Float16* __restrict__ hn) {
    int t = threadIdx.x, wv = t >> 5, lane = t & 31;
    size_t row = (size_t)blockIdx.x * 8 + wv;
    float4 v = ((const float4*)(x + row * 128))[lane];

    float ps = v.x + v.y + v.z + v.w;
    ps += __shfl_xor(ps, 1, 32);  ps += __shfl_xor(ps, 2, 32);
    ps += __shfl_xor(ps, 4, 32);  ps += __shfl_xor(ps, 8, 32);
    ps += __shfl_xor(ps, 16, 32);
    float mu = ps * (1.0f / 128.0f);

    float dx = v.x - mu, dy = v.y - mu, dz = v.z - mu, dw = v.w - mu;
    float pv = dx * dx + dy * dy + dz * dz + dw * dw;
    pv += __shfl_xor(pv, 1, 32);  pv += __shfl_xor(pv, 2, 32);
    pv += __shfl_xor(pv, 4, 32);  pv += __shfl_xor(pv, 8, 32);
    pv += __shfl_xor(pv, 16, 32);
    float rs = rsqrtf(pv * (1.0f / 128.0f) + 1e-5f);

    float4 w = ((const float4*)lnw)[lane];
    float4 b = ((const float4*)lnb)[lane];
    union { _Float16 h[4]; uint2 u; } o;
    o.h[0] = (_Float16)(dx * rs * w.x + b.x);
    o.h[1] = (_Float16)(dy * rs * w.y + b.y);
    o.h[2] = (_Float16)(dz * rs * w.z + b.z);
    o.h[3] = (_Float16)(dw * rs * w.w + b.w);
    *(uint2*)(hn + row * 128 + lane * 4) = o.u;
}

// ---------------------------------------------------------------------------
// Kernel 2: fused projections. Each block = one 16-row tile of hn.
// 33 column tiles of 16: 0..31 -> Q,K,V,G (f16 out); 32 -> tri_bias (f32 out).
// ---------------------------------------------------------------------------
__global__ __launch_bounds__(256) void proj_kernel(const _Float16* __restrict__ hn,
                                                   const _Float16* __restrict__ wcat,
                                                   _Float16* __restrict__ qkvg,
                                                   float* __restrict__ biasb) {
    int rt = blockIdx.x;                  // row tile 0..4095
    int t = threadIdx.x, wv = t >> 5, lane = t & 31;
    __shared__ _Float16 atile[16 * 128];  // 4 KB

    // async stage: 256 threads x 16B = 4KB
    async_copy_b128(lds_addr_of(atile) + t * 16, hn + (size_t)rt * 2048 + t * 8);
    wait_async0();
    __syncthreads();

    v16h a[4];
#pragma unroll
    for (int kc = 0; kc < 4; ++kc) a[kc] = load_frag16(atile + kc * 32, 128, lane);

    int n = lane & 15, hi = lane >> 4;
    for (int ct = wv; ct < 33; ct += 8) {
        const _Float16* wb = wcat + ct * 16 * 128;
        v16h b0 = load_frag16(wb,      128, lane);
        v16h b1 = load_frag16(wb + 32, 128, lane);
        v16h b2 = load_frag16(wb + 64, 128, lane);
        v16h b3 = load_frag16(wb + 96, 128, lane);
        v8f c = {};
        c = __builtin_amdgcn_wmma_f32_16x16x32_f16(false, a[0], false, b0, (short)0, c, false, false);
        c = __builtin_amdgcn_wmma_f32_16x16x32_f16(false, a[1], false, b1, (short)0, c, false, false);
        c = __builtin_amdgcn_wmma_f32_16x16x32_f16(false, a[2], false, b2, (short)0, c, false, false);
        c = __builtin_amdgcn_wmma_f32_16x16x32_f16(false, a[3], false, b3, (short)0, c, false, false);
        if (ct < 32) {
            int mat = ct >> 3, sub = ct & 7;
            _Float16* outp = qkvg + (size_t)mat * NN_ * 128;
            int col = sub * 16 + n;
#pragma unroll
            for (int r = 0; r < 8; ++r)
                outp[(size_t)(rt * 16 + r + hi * 8) * 128 + col] = (_Float16)c[r];
        } else if (n < 4) {   // bias tile: column n = head
#pragma unroll
            for (int r = 0; r < 8; ++r)
                biasb[(size_t)n * NN_ + rt * 16 + r + hi * 8] = c[r];
        }
    }
}

// ---------------------------------------------------------------------------
// Kernel 3: attention. One workgroup per (i, h); 8 waves; each wave owns two
// 16-row query tiles. Scores live in registers (16 x v8f = 128 VGPRs);
// softmax row-reductions are per-lane + 4 shfl_xor steps within the 16-lane
// half that shares rows (C-layout: lane = column). Only P (exp'd, f16) goes
// through LDS to transpose into the A-fragment layout for the P.V WMMAs.
// K and V staged with GLOBAL_LOAD_ASYNC_TO_LDS_B128; V transposed in LDS.
// LDS map (bytes):
//   0      Ksh  [256][32] f16  16384
//   16384  Vtmp [256][32] f16  16384   (async landing zone)
//   32768  Vsh  [32][256] f16  16384   (transposed for B-fragments)
//   49152  Psh  8 x [16][256] f16 65536  -> total 114688 (2 WGs per WGP)
// ---------------------------------------------------------------------------
__global__ __launch_bounds__(256) void attn_kernel(const _Float16* __restrict__ qkvg,
                                                   const float* __restrict__ biasb,
                                                   _Float16* __restrict__ og) {
    extern __shared__ char lds[];
    _Float16* Ksh  = (_Float16*)(lds);
    _Float16* Vtmp = (_Float16*)(lds + 16384);
    _Float16* Vsh  = (_Float16*)(lds + 32768);
    int i = blockIdx.x, h = blockIdx.y;
    int t = threadIdx.x, wv = t >> 5, lane = t & 31;
    _Float16* Psh = (_Float16*)(lds + 49152) + (size_t)wv * 16 * 256;

    const _Float16* Qg = qkvg;
    const _Float16* Kg = qkvg + (size_t)1 * NN_ * 128;
    const _Float16* Vg = qkvg + (size_t)2 * NN_ * 128;
    const _Float16* Gg = qkvg + (size_t)3 * NN_ * 128;

    // Async stage: thread t owns key row t (64B of K, 64B of V -> 4+4 b128).
    {
        const _Float16* kg = Kg + (size_t)(i * 256 + t) * 128 + h * 32;
        const _Float16* vg = Vg + (size_t)(i * 256 + t) * 128 + h * 32;
        uint32_t kdst = lds_addr_of(Ksh  + t * 32);
        uint32_t vdst = lds_addr_of(Vtmp + t * 32);
#pragma unroll
        for (int q = 0; q < 4; ++q) {
            async_copy_b128(kdst + q * 16, kg + q * 8);
            async_copy_b128(vdst + q * 16, vg + q * 8);
        }
    }
    wait_async0();
    __syncthreads();
    // Transpose V in LDS: Vsh[dh][key] = Vtmp[key][dh]
    for (int p = t; p < 256 * 32; p += 256) {
        int k = p >> 5, dh = p & 31;
        Vsh[dh * 256 + k] = Vtmp[k * 32 + dh];
    }
    __syncthreads();

    int n = lane & 15, hi = lane >> 4;
    for (int jt = wv * 2; jt < wv * 2 + 2; ++jt) {
        int jbase = jt * 16;
        // Q fragment (Q pre-scaled by 1/sqrt(DH) in proj)
        v16h aq = load_frag16(Qg + (size_t)(i * 256 + jbase) * 128 + h * 32, 128, lane);

        // ---- scores: S = Q K^T + bias (bias in accumulator), in registers ----
        v8f s[16];
        const float* bbase = biasb + (size_t)h * NN_ + (size_t)jbase * 256 + n;
#pragma unroll
        for (int kt = 0; kt < 16; ++kt) {
            v16h bk = load_frag16(Ksh + kt * 16 * 32, 32, lane);
            v8f c;
#pragma unroll
            for (int r = 0; r < 8; ++r) c[r] = bbase[(size_t)(r + hi * 8) * 256 + kt * 16];
            s[kt] = __builtin_amdgcn_wmma_f32_16x16x32_f16(false, aq, false, bk,
                                                           (short)0, c, false, false);
        }

        // ---- softmax: per-lane reduce over kt, then across 16 cols (shfl) ----
        float mx[8], sm[8];
#pragma unroll
        for (int r = 0; r < 8; ++r) {
            float m0 = s[0][r];
#pragma unroll
            for (int kt = 1; kt < 16; ++kt) m0 = fmaxf(m0, s[kt][r]);
            m0 = fmaxf(m0, __shfl_xor(m0, 1, 32));
            m0 = fmaxf(m0, __shfl_xor(m0, 2, 32));
            m0 = fmaxf(m0, __shfl_xor(m0, 4, 32));
            m0 = fmaxf(m0, __shfl_xor(m0, 8, 32));
            mx[r] = m0;
            sm[r] = 0.0f;
        }
#pragma unroll
        for (int kt = 0; kt < 16; ++kt) {
#pragma unroll
            for (int r = 0; r < 8; ++r) {
                float e = __expf(s[kt][r] - mx[r]);
                sm[r] += e;
                Psh[(r + hi * 8) * 256 + kt * 16 + n] = (_Float16)e;
            }
        }
#pragma unroll
        for (int r = 0; r < 8; ++r) {
            float v = sm[r];
            v += __shfl_xor(v, 1, 32);
            v += __shfl_xor(v, 2, 32);
            v += __shfl_xor(v, 4, 32);
            v += __shfl_xor(v, 8, 32);
            sm[r] = v;   // row sum for row (r + hi*8) -- same row map as O accum
        }
        __syncthreads();   // P writes -> P fragment reads (uniform per wave)

        // ---- O = P V (two dh tiles of 16) ----
        v8f o0 = {}, o1 = {};
#pragma unroll
        for (int kp = 0; kp < 8; ++kp) {
            v16h ap  = load_frag16(Psh + kp * 32, 256, lane);
            v16h bv0 = load_frag16(Vsh + kp * 32, 256, lane);
            v16h bv1 = load_frag16(Vsh + 16 * 256 + kp * 32, 256, lane);
            o0 = __builtin_amdgcn_wmma_f32_16x16x32_f16(false, ap, false, bv0,
                                                        (short)0, o0, false, false);
            o1 = __builtin_amdgcn_wmma_f32_16x16x32_f16(false, ap, false, bv1,
                                                        (short)0, o1, false, false);
        }

        // ---- normalize, gate with sigmoid(g), store head-major ----
#pragma unroll
        for (int r = 0; r < 8; ++r) {
            float inv = 1.0f / sm[r];
            size_t row = (size_t)(i * 256 + jbase + r + hi * 8);
            float g0 = (float)Gg[row * 128 + h * 32 + n];
            float g1 = (float)Gg[row * 128 + h * 32 + 16 + n];
            float s0 = 1.0f / (1.0f + __expf(-g0));
            float s1 = 1.0f / (1.0f + __expf(-g1));
            og[row * 128 + h * 32 + n]      = (_Float16)(o0[r] * inv * s0);
            og[row * 128 + h * 32 + 16 + n] = (_Float16)(o1[r] * inv * s1);
        }
    }
}

// ---------------------------------------------------------------------------
// Kernel 4: output projection out = og @ o_w^T  (f32 result)
// ---------------------------------------------------------------------------
__global__ __launch_bounds__(256) void out_kernel(const _Float16* __restrict__ og,
                                                  const _Float16* __restrict__ wo,
                                                  float* __restrict__ out) {
    int rt = blockIdx.x, t = threadIdx.x;
    int wv = t >> 5, lane = t & 31;
    __shared__ _Float16 atile[16 * 128];

    async_copy_b128(lds_addr_of(atile) + t * 16, og + (size_t)rt * 2048 + t * 8);
    wait_async0();
    __syncthreads();

    v16h a[4];
#pragma unroll
    for (int kc = 0; kc < 4; ++kc) a[kc] = load_frag16(atile + kc * 32, 128, lane);

    int ct = wv;           // 8 waves x 8 column tiles
    const _Float16* wb = wo + ct * 16 * 128;
    v16h b0 = load_frag16(wb,      128, lane);
    v16h b1 = load_frag16(wb + 32, 128, lane);
    v16h b2 = load_frag16(wb + 64, 128, lane);
    v16h b3 = load_frag16(wb + 96, 128, lane);
    v8f c = {};
    c = __builtin_amdgcn_wmma_f32_16x16x32_f16(false, a[0], false, b0, (short)0, c, false, false);
    c = __builtin_amdgcn_wmma_f32_16x16x32_f16(false, a[1], false, b1, (short)0, c, false, false);
    c = __builtin_amdgcn_wmma_f32_16x16x32_f16(false, a[2], false, b2, (short)0, c, false, false);
    c = __builtin_amdgcn_wmma_f32_16x16x32_f16(false, a[3], false, b3, (short)0, c, false, false);

    int hi = lane >> 4, n = lane & 15;
#pragma unroll
    for (int r = 0; r < 8; ++r)
        out[(size_t)(rt * 16 + r + hi * 8) * 128 + ct * 16 + n] = c[r];
}

// ---------------------------------------------------------------------------
extern "C" void kernel_launch(void* const* d_in, const int* in_sizes, int n_in,
                              void* d_out, int out_size, void* d_ws, size_t ws_size,
                              hipStream_t stream) {
    const float* x   = (const float*)d_in[0];
    const float* lnw = (const float*)d_in[1];
    const float* lnb = (const float*)d_in[2];
    const float* bw  = (const float*)d_in[3];
    const float* qw  = (const float*)d_in[4];
    const float* kw  = (const float*)d_in[5];
    const float* vw  = (const float*)d_in[6];
    const float* gw  = (const float*)d_in[7];
    const float* ow  = (const float*)d_in[8];
    float* out = (float*)d_out;

    char* ws = (char*)d_ws;
    _Float16* hn    = (_Float16*)ws;                       // NN*D f16
    _Float16* qkvg  = hn + (size_t)NN_ * D_;               // 4*NN*D f16
    float*    biasb = (float*)(qkvg + (size_t)4 * NN_ * D_);  // H*NN f32
    _Float16* og    = (_Float16*)(biasb + (size_t)H_ * NN_);  // NN*D f16
    _Float16* wcat  = og + (size_t)NN_ * D_;               // 528*128 f16
    _Float16* wo    = wcat + 528 * 128;                    // 128*128 f16

    prep_weights<<<328, 256, 0, stream>>>(qw, kw, vw, gw, bw, ow, wcat, wo);
    ln_kernel<<<NN_ / 8, 256, 0, stream>>>(x, lnw, lnb, hn);
    proj_kernel<<<NN_ / 16, 256, 0, stream>>>(hn, wcat, qkvg, biasb);
    attn_kernel<<<dim3(256, 4), 256, 114688, stream>>>(qkvg, biasb, og);
    out_kernel<<<NN_ / 16, 256, 0, stream>>>(og, wo, out);
}
